// RNN_14362370638204
// MI455X (gfx1250) — compile-verified, matrix operand order checked by reading, not run on hardware
//
#include <hip/hip_runtime.h>
#include <math.h>

#define SEQ     2048
#define BATCH   256
#define IN_DIM  128
#define HID     256
#define NCLS    10

// LDS row strides in bf16 (ushort) units; +8 pad breaks 64-bank conflicts.
#define XT_STRIDE  136   // 128 + 8
#define HS_STRIDE  264   // 256 + 8

typedef __attribute__((ext_vector_type(16))) __bf16 v16bf;
typedef __attribute__((ext_vector_type(8)))  __bf16 v8bf;
typedef __attribute__((ext_vector_type(8)))  float  v8f;

union FragU  { v16bf v; v8bf h[2]; };
union FragB  { unsigned short s[16]; v16bf v; };
union Pack8  { unsigned short s[8]; uint4 u4; };

__device__ __forceinline__ unsigned short f2bf_bits(float f) {
  union { float f; unsigned int u; } c; c.f = f;
  unsigned int u = c.u + 0x7FFFu + ((c.u >> 16) & 1u);   // round-to-nearest-even
  return (unsigned short)(u >> 16);
}

__device__ __forceinline__ float fast_tanh(float x) {
#if __has_builtin(__builtin_amdgcn_tanhf)
  return __builtin_amdgcn_tanhf(x);       // v_tanh_f32 (gfx1250 transcendental)
#elif __has_builtin(__builtin_amdgcn_tanh_f32)
  return __builtin_amdgcn_tanh_f32(x);
#else
  return tanhf(x);
#endif
}

// B-operand fragment from row-major [N][K] LDS (bf16). Per CDNA5 ISA 16-bit
// layout: lanes 0-15 hold K = {k0..+7, k0+16..+23}; lanes 16-31 K = {k0+8..+15, k0+24..+31}.
__device__ __forceinline__ v16bf frag_ld(const unsigned short* rowp, int k0, int half) {
  FragU u;
  u.h[0] = *(const v8bf*)(rowp + k0 + half * 8);
  u.h[1] = *(const v8bf*)(rowp + k0 + 16 + half * 8);
  return u.v;
}

// A-operand fragment built once from global f32 weights (same K mapping, M = lane&15 row).
__device__ __forceinline__ v16bf fragA_global(const float* rowp, int k0, int half) {
  FragB u;
#pragma unroll
  for (int i = 0; i < 8; ++i) {
    u.s[i]     = f2bf_bits(rowp[k0 + half * 8 + i]);
    u.s[8 + i] = f2bf_bits(rowp[k0 + 16 + half * 8 + i]);
  }
  return u.v;
}

// One workgroup (256 thr = 8 waves) owns a 16-batch slice for all 2048 steps.
// Weights live in REGISTERS (loop-invariant A-fragments); h and x tiles
// double-buffer in LDS. Fused: h = tanh(x_t@W_ih^T + h@W_hh^T + b).
__global__ __launch_bounds__(256, 1) void rnn_recurrence_kernel(
    const float* __restrict__ x, const float* __restrict__ W_ih,
    const float* __restrict__ W_hh, const float* __restrict__ b_ih,
    const float* __restrict__ b_hh, float* __restrict__ h_last) {
  __shared__ __attribute__((aligned(16))) unsigned short s_x[2][16 * XT_STRIDE];
  __shared__ __attribute__((aligned(16))) unsigned short s_h[2][16 * HS_STRIDE];

  const int tid    = threadIdx.x;
  const int lane   = tid & 31;
  const int half   = (lane >> 4) & 1;
  const int wave   = tid >> 5;
  const int bl_row = lane & 15;          // local batch row
  const int b0     = blockIdx.x * 16;    // global batch base
  const int row    = tid >> 4;           // x-tile staging row for this thread
  const int col    = (tid & 15) * 8;     // x-tile staging col (8 floats)

  // ---- one-time init ----
  // h0 = 0 (buffer 0 only; buffer 1 is fully written in step 0 before any read)
  for (int i = tid; i < 16 * HS_STRIDE; i += 256) s_h[0][i] = 0;
  {  // stage x[t=0] tile as bf16
    const float* src = x + (size_t)(b0 + row) * IN_DIM + col;
    Pack8 p;
#pragma unroll
    for (int i = 0; i < 8; ++i) p.s[i] = f2bf_bits(src[i]);
    *(uint4*)&s_x[0][row * XT_STRIDE + col] = p.u4;
  }

  // Loop-invariant A-fragments and bias -> registers.
  // Wave w owns j-tiles {w, w+8}; each tile: 4 K-chunks over IN, 8 over HID.
  v16bf a_ih[2][4], a_hh[2][8];
  float biasr[2][8];
#pragma unroll
  for (int jj = 0; jj < 2; ++jj) {
    const int j0 = (wave + jj * 8) * 16;
    const float* rih = W_ih + (size_t)(j0 + bl_row) * IN_DIM;
    const float* rhh = W_hh + (size_t)(j0 + bl_row) * HID;
#pragma unroll
    for (int kk = 0; kk < 4; ++kk) a_ih[jj][kk] = fragA_global(rih, kk * 32, half);
#pragma unroll
    for (int kk = 0; kk < 8; ++kk) a_hh[jj][kk] = fragA_global(rhh, kk * 32, half);
#pragma unroll
    for (int r = 0; r < 8; ++r) {
      const int j = j0 + half * 8 + r;   // D-tile: M = r + half*8
      biasr[jj][r] = b_ih[j] + b_hh[j];
    }
  }
  __syncthreads();

  // ---- 2048 serial steps ----
  for (int t = 0; t < SEQ; ++t) {
    const int cur = t & 1, nxt = cur ^ 1;

    // register-prefetch x[t+1] tile (overlaps WMMAs); L2-prefetch x[t+2]
    float pf[8];
    const bool has_next = (t + 1 < SEQ);
    if (has_next) {
      const float* src = x + ((size_t)(t + 1) * BATCH + b0 + row) * IN_DIM + col;
      const float4 a = ((const float4*)src)[0];
      const float4 b = ((const float4*)src)[1];
      pf[0] = a.x; pf[1] = a.y; pf[2] = a.z; pf[3] = a.w;
      pf[4] = b.x; pf[5] = b.y; pf[6] = b.z; pf[7] = b.w;
    }
    if (t + 2 < SEQ)
      __builtin_prefetch(x + ((size_t)(t + 2) * BATCH + b0 + row) * IN_DIM + col, 0, 0);

    // GEMM over 12 K-chunks (4 from x@W_ih^T, 8 from h@W_hh^T).
    // Software-pipelined: issue B-fragment load for chunk k+1 BEFORE the
    // chunk-k WMMAs so the LDS latency hides under XDL execution
    // (compiler can leave the next loads outstanding across the wait).
    const unsigned short* brow_x = s_x[cur] + bl_row * XT_STRIDE;
    const unsigned short* brow_h = s_h[cur] + bl_row * HS_STRIDE;
    v8f acc0 = {}, acc1 = {};
    v16bf b_cur = frag_ld(brow_x, 0, half);
#pragma unroll
    for (int kk = 0; kk < 12; ++kk) {
      v16bf b_nxt = b_cur;
      if (kk + 1 < 12)
        b_nxt = (kk + 1 < 4) ? frag_ld(brow_x, (kk + 1) * 32, half)
                             : frag_ld(brow_h, (kk - 3) * 32, half);
      const v16bf a0f = (kk < 4) ? a_ih[0][kk] : a_hh[0][kk - 4];
      const v16bf a1f = (kk < 4) ? a_ih[1][kk] : a_hh[1][kk - 4];
      acc0 = __builtin_amdgcn_wmma_f32_16x16x32_bf16(false, a0f, false, b_cur,
                                                     (short)0, acc0, false, false);
      acc1 = __builtin_amdgcn_wmma_f32_16x16x32_bf16(false, a1f, false, b_cur,
                                                     (short)0, acc1, false, false);
      b_cur = b_nxt;
    }

    // epilogue: tanh + one packed b128 store per tile (8 consecutive j slots)
#pragma unroll
    for (int jj = 0; jj < 2; ++jj) {
      const int j0 = (wave + jj * 8) * 16;
      const v8f acc = jj ? acc1 : acc0;
      Pack8 p;
#pragma unroll
      for (int r = 0; r < 8; ++r) {
        const float v = fast_tanh(acc[r] + biasr[jj][r]);
        p.s[r] = f2bf_bits(v);
        if (t == SEQ - 1)
          h_last[(size_t)(b0 + bl_row) * HID + j0 + half * 8 + r] = v;  // f32 state for head
      }
      *(uint4*)&s_h[nxt][bl_row * HS_STRIDE + j0 + half * 8] = p.u4;
    }

    if (has_next) {                        // stage x[t+1] as bf16 (one b128 store)
      Pack8 p;
#pragma unroll
      for (int i = 0; i < 8; ++i) p.s[i] = f2bf_bits(pf[i]);
      *(uint4*)&s_x[nxt][row * XT_STRIDE + col] = p.u4;
    }
    __syncthreads();   // separates step t (reads cur / writes nxt) from step t+1
  }
}

// FC head + softmax over the BATCH axis (reference softmax(dim=1) on (1,B,C)).
__global__ __launch_bounds__(256, 1) void rnn_head_kernel(
    const float* __restrict__ h_last, const float* __restrict__ W_fc,
    const float* __restrict__ b_fc, float* __restrict__ out) {
  __shared__ float s_wfc[NCLS * HID];
  __shared__ float s_logit[BATCH][NCLS];
  __shared__ float s_mx[NCLS], s_sum[NCLS];
  const int tid = threadIdx.x;   // one thread per batch row
  for (int i = tid; i < NCLS * HID; i += 256) s_wfc[i] = W_fc[i];
  __syncthreads();

  float acc[NCLS];
#pragma unroll
  for (int c = 0; c < NCLS; ++c) acc[c] = b_fc[c];
  for (int k = 0; k < HID; ++k) {
    const float hv = h_last[(size_t)tid * HID + k];
#pragma unroll
    for (int c = 0; c < NCLS; ++c) acc[c] += hv * s_wfc[c * HID + k];
  }
#pragma unroll
  for (int c = 0; c < NCLS; ++c) s_logit[tid][c] = acc[c];
  __syncthreads();

  if (tid < NCLS) {            // reduce over the batch axis per class
    float mx = -INFINITY;
    for (int b = 0; b < BATCH; ++b) mx = fmaxf(mx, s_logit[b][tid]);
    float sm = 0.f;
    for (int b = 0; b < BATCH; ++b) sm += expf(s_logit[b][tid] - mx);
    s_mx[tid] = mx; s_sum[tid] = sm;
  }
  __syncthreads();
#pragma unroll
  for (int c = 0; c < NCLS; ++c)
    out[(size_t)tid * NCLS + c] = expf(s_logit[tid][c] - s_mx[c]) / s_sum[c];
}

extern "C" void kernel_launch(void* const* d_in, const int* in_sizes, int n_in,
                              void* d_out, int out_size, void* d_ws, size_t ws_size,
                              hipStream_t stream) {
  const float* x    = (const float*)d_in[0];
  const float* W_ih = (const float*)d_in[1];
  const float* W_hh = (const float*)d_in[2];
  const float* b_ih = (const float*)d_in[3];
  const float* b_hh = (const float*)d_in[4];
  const float* W_fc = (const float*)d_in[5];
  const float* b_fc = (const float*)d_in[6];

  float* h_last = (float*)d_ws;   // BATCH*HID f32 = 256 KB scratch

  rnn_recurrence_kernel<<<BATCH / 16, 256, 0, stream>>>(x, W_ih, W_hh, b_ih, b_hh, h_last);
  rnn_head_kernel<<<1, 256, 0, stream>>>(h_last, W_fc, b_fc, (float*)d_out);
}
